// LayerMultiheadSelfAttention_60979945669022
// MI455X (gfx1250) — matrix operand order repaired and checked by reading, not compiled, not run on hardware
//
#include <hip/hip_runtime.h>
#include <hip/hip_bf16.h>

// ---------------------------------------------------------------------------
// Fused multi-head self-attention block for gfx1250 (MI455X), wave32 + WMMA.
//   B=4, S=2048, F=256, H=8, D=32
// Pipeline:
//   0) prep_weights  : Wq/Wk/Wv/Wf f32[K][N] -> f16 transposed [N][K]
//   1) qkv_proj      : single pass over x; 6 WMMAs per K-chunk (Q,K,V fused)
//   2) attn          : flash-attention, 64-key chunks (8 WMMAs per softmax
//                      update to amortize the cross-lane reductions)
//   3) out_proj_ln   : ctx@Wf + bias, ReLU, +x residual, LayerNorm
// ---------------------------------------------------------------------------

typedef _Float16 h16;
typedef __attribute__((ext_vector_type(16))) _Float16 v16h;
typedef __attribute__((ext_vector_type(8)))  _Float16 v8h;
typedef __attribute__((ext_vector_type(8)))  float    v8f;

constexpr int BATCH = 4;
constexpr int SEQ   = 2048;
constexpr int FDIM  = 256;
constexpr int NH    = 8;
constexpr int HD    = 32;
constexpr int ROWS  = BATCH * SEQ;                   // 8192
constexpr int WPW   = 4;                             // waves/WG (attention)
constexpr int PSTR  = 72;                            // P staging row stride
constexpr size_t WT_ELEMS  = (size_t)FDIM * FDIM;    // 65536 per matrix
constexpr size_t QKV_ELEMS = (size_t)ROWS * FDIM;    // 2M half per tensor

__device__ __forceinline__ v8f wmma16(v16h a, v16h b, v8f c) {
  // (neg_a, A, neg_b, B, c_mod, C, reuse_a, reuse_b)
  return __builtin_amdgcn_wmma_f32_16x16x32_f16(false, a, false, b, (short)0, c,
                                                false, false);
}

// 16 contiguous f16 -> one fragment
__device__ __forceinline__ v16h ld16(const h16* __restrict__ p) {
  v8h lo = *(const v8h*)p;
  v8h hi = *(const v8h*)(p + 8);
  v16h r;
#pragma unroll
  for (int i = 0; i < 8; ++i) { r[i] = lo[i]; r[8 + i] = hi[i]; }
  return r;
}

// two separate 8-element halves (A-fragment K-split layout)
__device__ __forceinline__ v16h ld8x2(const h16* __restrict__ p0,
                                      const h16* __restrict__ p1) {
  v8h lo = *(const v8h*)p0;
  v8h hi = *(const v8h*)p1;
  v16h r;
#pragma unroll
  for (int i = 0; i < 8; ++i) { r[i] = lo[i]; r[8 + i] = hi[i]; }
  return r;
}

// ---------------------------------------------------------------------------
// Kernel 0: convert + transpose weights:  W f32[K][N] -> WT f16[N][K].
// ---------------------------------------------------------------------------
__global__ __launch_bounds__(256) void prep_weights_kernel(
    const float* __restrict__ Wq, const float* __restrict__ Wk,
    const float* __restrict__ Wv, const float* __restrict__ Wf,
    h16* __restrict__ WT) {
  const int m = blockIdx.y;
  const float* W = (m == 0) ? Wq : (m == 1) ? Wk : (m == 2) ? Wv : Wf;
  h16* dst = WT + (size_t)m * WT_ELEMS;
  const int k = blockIdx.x;
  const int n = threadIdx.x;
  dst[(size_t)n * FDIM + k] = (h16)W[(size_t)k * FDIM + n];
}

// ---------------------------------------------------------------------------
// Kernel 1: fused QKV projection.  grid = ROWS/16, block = 256 (8 waves).
// Wave w owns columns [w*32, w*32+32) == head w.  One pass over x feeds
// 6 accumulators (Q/K/V x 2 n-blocks): 6 WMMAs per 32-wide K-chunk.
// Q pre-scaled by 1/sqrt(H); V stored transposed Vt[b,h,d,s].
// ---------------------------------------------------------------------------
__global__ __launch_bounds__(256) void qkv_proj_kernel(
    const float* __restrict__ x,
    const h16* __restrict__ WqT, const h16* __restrict__ WkT,
    const h16* __restrict__ WvT,
    h16* __restrict__ Qs, h16* __restrict__ Ks, h16* __restrict__ Vt) {
  const int lane   = threadIdx.x & 31;
  const int wave   = threadIdx.x >> 5;       // == head index
  const int rowblk = blockIdx.x;             // 16-row block of [B*S]

  const int arow  = lane & 15;               // A row within tile
  const int kbase = (lane >> 4) * 8;         // A K-split base
  const int cc    = lane & 15;               // B column within tile
  const int bk    = (lane >> 4) * 16;        // B K base
  const int n0    = wave * 32;               // output column base

  v8f cq0 = {}, cq1 = {}, ck0 = {}, ck1 = {}, cv0 = {}, cv1 = {};
  const float* xrow = x + ((size_t)rowblk * 16 + arow) * FDIM;
  const h16* wq0 = WqT + (size_t)(n0 + cc) * FDIM;
  const h16* wq1 = WqT + (size_t)(n0 + 16 + cc) * FDIM;
  const h16* wk0 = WkT + (size_t)(n0 + cc) * FDIM;
  const h16* wk1 = WkT + (size_t)(n0 + 16 + cc) * FDIM;
  const h16* wv0 = WvT + (size_t)(n0 + cc) * FDIM;
  const h16* wv1 = WvT + (size_t)(n0 + 16 + cc) * FDIM;

#pragma unroll
  for (int kc = 0; kc < FDIM / 32; ++kc) {
    v16h a;
    const float* ap = xrow + kc * 32 + bk;
#pragma unroll
    for (int e = 0; e < 8; ++e) a[e] = (h16)ap[e];
#pragma unroll
    for (int e = 0; e < 8; ++e) a[8 + e] = (h16)ap[8 + e];
    const int ko = kc * 32 + bk;           // contiguous along K in WT[N][K]
    cq0 = wmma16(a, ld16(wq0 + ko), cq0);
    cq1 = wmma16(a, ld16(wq1 + ko), cq1);
    ck0 = wmma16(a, ld16(wk0 + ko), ck0);
    ck1 = wmma16(a, ld16(wk1 + ko), ck1);
    cv0 = wmma16(a, ld16(wv0 + ko), cv0);
    cv1 = wmma16(a, ld16(wv1 + ko), cv1);
  }

  const int   mbase = (lane >> 4) * 8;
  const float qsc   = 0.35355339059327373f;   // 1/sqrt(H)
  const int   h     = wave;
#pragma unroll
  for (int r = 0; r < 8; ++r) {
    const int gr = rowblk * 16 + mbase + r;       // global row in [B*S]
    const int b  = gr / SEQ, s = gr % SEQ;
    const size_t qkbase = (((size_t)b * NH + h) * SEQ + s) * HD;
    Qs[qkbase + cc]      = (h16)(cq0[r] * qsc);
    Qs[qkbase + cc + 16] = (h16)(cq1[r] * qsc);
    Ks[qkbase + cc]      = (h16)ck0[r];
    Ks[qkbase + cc + 16] = (h16)ck1[r];
    Vt[(((size_t)b * NH + h) * HD + cc) * SEQ + s]        = (h16)cv0[r];
    Vt[(((size_t)b * NH + h) * HD + (cc + 16)) * SEQ + s] = (h16)cv1[r];
  }
}

// ---------------------------------------------------------------------------
// Kernel 2: flash attention, 64-key chunks.  One wave owns a 16-row Q tile.
// Per chunk: 4 score WMMAs -> one online-softmax update (max/sum reductions
// amortized over 64 keys) -> P relayout via LDS -> 4 PV WMMAs.
// ---------------------------------------------------------------------------
__global__ __launch_bounds__(32 * WPW) void attn_kernel(
    const h16* __restrict__ Q,   // [B*H, S, 32]  (pre-scaled)
    const h16* __restrict__ K,   // [B*H, S, 32]
    const h16* __restrict__ Vt,  // [B*H, 32, S]
    h16* __restrict__ ctx) {     // [B*S, 256]  merged-head
  __shared__ h16 pstage[WPW][16][PSTR];   // 16x64 P tile, padded row stride

  const int lane = threadIdx.x & 31;
  const int wave = threadIdx.x >> 5;
  const int tile = blockIdx.x * WPW + wave;   // 16-row tile id
  const int bh   = tile / (SEQ / 16);
  const int srow = (tile % (SEQ / 16)) * 16;
  const int b    = bh / NH, h = bh % NH;

  const h16* Qp = Q  + ((size_t)bh * SEQ + srow) * HD;
  const h16* Kp = K  + (size_t)bh * SEQ * HD;
  const h16* Vp = Vt + (size_t)bh * HD * SEQ;

  const int arow  = lane & 15;
  const int kbase = (lane >> 4) * 8;
  const int cc    = lane & 15;
  const int bk    = (lane >> 4) * 16;
  const int mbase = (lane >> 4) * 8;

  // Q A-fragment: 16 rows x 32 (= full head dim) — 8 VGPRs, loop-invariant
  const v16h qa = ld16(Qp + (size_t)arow * HD + bk);

  v8f acc0 = {}, acc1 = {};
  float m_run[8], l_run[8];
#pragma unroll
  for (int r = 0; r < 8; ++r) { m_run[r] = -1e30f; l_run[r] = 0.0f; }
  const float L2E = 1.44269504088896340736f;

  for (int j = 0; j < SEQ / 64; ++j) {
    const int k0 = j * 64;
    // 4 score fragments covering keys [k0, k0+64)
    v8f s0 = {}, s1 = {}, s2 = {}, s3 = {};
    s0 = wmma16(qa, ld16(Kp + (size_t)(k0 + cc) * HD + bk), s0);
    s1 = wmma16(qa, ld16(Kp + (size_t)(k0 + 16 + cc) * HD + bk), s1);
    s2 = wmma16(qa, ld16(Kp + (size_t)(k0 + 32 + cc) * HD + bk), s2);
    s3 = wmma16(qa, ld16(Kp + (size_t)(k0 + 48 + cc) * HD + bk), s3);

    // one online-softmax update for all 64 keys
#pragma unroll
    for (int r = 0; r < 8; ++r) {
      float mx = fmaxf(fmaxf(s0[r], s1[r]), fmaxf(s2[r], s3[r]));
      mx = fmaxf(mx, __shfl_xor(mx, 1, 32));
      mx = fmaxf(mx, __shfl_xor(mx, 2, 32));
      mx = fmaxf(mx, __shfl_xor(mx, 4, 32));
      mx = fmaxf(mx, __shfl_xor(mx, 8, 32));
      const float m_new = fmaxf(m_run[r], mx);
      const float a  = __builtin_exp2f((m_run[r] - m_new) * L2E);
      const float p0 = __builtin_exp2f((s0[r] - m_new) * L2E);
      const float p1 = __builtin_exp2f((s1[r] - m_new) * L2E);
      const float p2 = __builtin_exp2f((s2[r] - m_new) * L2E);
      const float p3 = __builtin_exp2f((s3[r] - m_new) * L2E);
      float rs = (p0 + p1) + (p2 + p3);
      rs += __shfl_xor(rs, 1, 32);
      rs += __shfl_xor(rs, 2, 32);
      rs += __shfl_xor(rs, 4, 32);
      rs += __shfl_xor(rs, 8, 32);
      l_run[r] = l_run[r] * a + rs;
      m_run[r] = m_new;
      s0[r] = p0; s1[r] = p1; s2[r] = p2; s3[r] = p3;
      acc0[r] *= a; acc1[r] *= a;
    }

    // relayout P: C-fragment (row across lanes) -> A-fragment (row per lane)
    h16 (*st)[PSTR] = pstage[wave];
#pragma unroll
    for (int r = 0; r < 8; ++r) {
      st[mbase + r][cc]      = (h16)s0[r];
      st[mbase + r][cc + 16] = (h16)s1[r];
      st[mbase + r][cc + 32] = (h16)s2[r];
      st[mbase + r][cc + 48] = (h16)s3[r];
    }
    asm volatile("s_wait_dscnt 0" ::: "memory");  // same-wave LDS RAW fence
    const v16h pa0 = ld16(&st[arow][bk]);
    const v16h pa1 = ld16(&st[arow][bk + 32]);

    // PV: B-fragments from Vt (contiguous along s)
    acc0 = wmma16(pa0, ld16(Vp + (size_t)cc * SEQ + k0 + bk), acc0);
    acc0 = wmma16(pa1, ld16(Vp + (size_t)cc * SEQ + k0 + 32 + bk), acc0);
    acc1 = wmma16(pa0, ld16(Vp + (size_t)(cc + 16) * SEQ + k0 + bk), acc1);
    acc1 = wmma16(pa1, ld16(Vp + (size_t)(cc + 16) * SEQ + k0 + 32 + bk), acc1);
  }

  // epilogue: normalize by row sum, store merged-head ctx
#pragma unroll
  for (int r = 0; r < 8; ++r) {
    const float inv = 1.0f / l_run[r];
    const int   sg  = srow + mbase + r;
    const size_t row = (size_t)b * SEQ + sg;
    ctx[row * FDIM + h * HD + cc]      = (h16)(acc0[r] * inv);
    ctx[row * FDIM + h * HD + 16 + cc] = (h16)(acc1[r] * inv);
  }
}

// ---------------------------------------------------------------------------
// Kernel 3: out projection + bias + ReLU + residual + LayerNorm(no affine).
// grid = ROWS/16, block = 256 (8 waves); wave w owns columns [w*32, w*32+32).
// ---------------------------------------------------------------------------
__global__ __launch_bounds__(256) void out_proj_ln_kernel(
    const h16* __restrict__ ctx, const h16* __restrict__ WfT,
    const float* __restrict__ bias, const float* __restrict__ x,
    float* __restrict__ out) {
  __shared__ float zt[16][FDIM + 1];

  const int lane   = threadIdx.x & 31;
  const int wave   = threadIdx.x >> 5;
  const int rowblk = blockIdx.x;

  const int arow  = lane & 15;
  const int kbase = (lane >> 4) * 8;
  const int cc    = lane & 15;
  const int bk    = (lane >> 4) * 16;
  const int n0    = wave * 32;
  const int mbase = (lane >> 4) * 8;

  v8f c0 = {}, c1 = {};
  const h16* crow = ctx + ((size_t)rowblk * 16 + arow) * FDIM;
  const h16* wf0  = WfT + (size_t)(n0 + cc) * FDIM;
  const h16* wf1  = WfT + (size_t)(n0 + 16 + cc) * FDIM;

#pragma unroll
  for (int kc = 0; kc < FDIM / 32; ++kc) {
    const v16h a = ld16(crow + kc * 32 + bk);
    const int ko = kc * 32 + bk;
    c0 = wmma16(a, ld16(wf0 + ko), c0);
    c1 = wmma16(a, ld16(wf1 + ko), c1);
  }

  // bias + ReLU + residual -> LDS staging for the row-wise LayerNorm
  const float b0v = bias[n0 + cc];
  const float b1v = bias[n0 + 16 + cc];
#pragma unroll
  for (int r = 0; r < 8; ++r) {
    const size_t gr = (size_t)rowblk * 16 + mbase + r;
    float y0 = fmaxf(c0[r] + b0v, 0.0f) + x[gr * FDIM + n0 + cc];
    float y1 = fmaxf(c1[r] + b1v, 0.0f) + x[gr * FDIM + n0 + 16 + cc];
    zt[mbase + r][n0 + cc]      = y0;
    zt[mbase + r][n0 + 16 + cc] = y1;
  }
  __syncthreads();

  // LayerNorm (center=False, scale=False): 16 threads per row
  const int row = threadIdx.x >> 4;
  const int c16 = threadIdx.x & 15;
  float s = 0.0f, s2 = 0.0f;
#pragma unroll
  for (int i = 0; i < 16; ++i) {
    const float v = zt[row][c16 + 16 * i];
    s += v; s2 += v * v;
  }
  s  += __shfl_xor(s, 1, 32);  s  += __shfl_xor(s, 2, 32);
  s  += __shfl_xor(s, 4, 32);  s  += __shfl_xor(s, 8, 32);
  s2 += __shfl_xor(s2, 1, 32); s2 += __shfl_xor(s2, 2, 32);
  s2 += __shfl_xor(s2, 4, 32); s2 += __shfl_xor(s2, 8, 32);
  const float mean = s * (1.0f / FDIM);
  const float var  = s2 * (1.0f / FDIM) - mean * mean;
  const float rstd = rsqrtf(var + 1e-3f);
  const size_t gr = (size_t)rowblk * 16 + row;
#pragma unroll
  for (int i = 0; i < 16; ++i) {
    out[gr * FDIM + c16 + 16 * i] = (zt[row][c16 + 16 * i] - mean) * rstd;
  }
}

// ---------------------------------------------------------------------------
extern "C" void kernel_launch(void* const* d_in, const int* in_sizes, int n_in,
                              void* d_out, int out_size, void* d_ws,
                              size_t ws_size, hipStream_t stream) {
  const float* x    = (const float*)d_in[0];
  const float* Wq   = (const float*)d_in[1];
  const float* Wk   = (const float*)d_in[2];
  const float* Wv   = (const float*)d_in[3];
  const float* Wf   = (const float*)d_in[4];
  const float* bias = (const float*)d_in[5];
  float* out = (float*)d_out;

  char* ws = (char*)d_ws;
  h16* WT  = (h16*)(ws);                                   // 4 x 128 KB
  h16* Qs  = (h16*)(ws + WT_ELEMS * 8);                    // 4 MB
  h16* Ks  = (h16*)(ws + WT_ELEMS * 8 + QKV_ELEMS * 2);    // 4 MB
  h16* Vt  = (h16*)(ws + WT_ELEMS * 8 + QKV_ELEMS * 4);    // 4 MB
  h16* Ctx = (h16*)(ws + WT_ELEMS * 8 + QKV_ELEMS * 6);    // 4 MB
  h16* WqT = WT;
  h16* WkT = WT + WT_ELEMS;
  h16* WvT = WT + WT_ELEMS * 2;
  h16* WfT = WT + WT_ELEMS * 3;

  prep_weights_kernel<<<dim3(FDIM, 4), FDIM, 0, stream>>>(Wq, Wk, Wv, Wf, WT);
  qkv_proj_kernel<<<dim3(ROWS / 16), 256, 0, stream>>>(x, WqT, WkT, WvT,
                                                       Qs, Ks, Vt);
  attn_kernel<<<dim3((BATCH * NH * (SEQ / 16)) / WPW), 32 * WPW, 0, stream>>>(
      Qs, Ks, Vt, Ctx);
  out_proj_ln_kernel<<<dim3(ROWS / 16), 256, 0, stream>>>(Ctx, WfT, bias, x,
                                                          out);
}
